// ReconNet_23682449670350
// MI455X (gfx1250) — matrix operand order repaired and checked by reading, not compiled, not run on hardware
//
#include <hip/hip_runtime.h>
#include <hip/hip_bf16.h>

// ---------------------------------------------------------------------------
// MI455X (gfx1250) implementation of ReconNet.
// All heavy math (spatial GEMM layers, 9x9 spectral conv, 1x1 conv) runs on
// v_wmma_f32_16x16x32_bf16. Activations are stored K-contiguous so WMMA B
// fragments are single 32B contiguous reads per lane; weights are pre-packed
// bf16. The 9x9 conv stages its weight slab + image rows into LDS with
// GLOBAL_LOAD_ASYNC_TO_LDS_B128 (ASYNCcnt) and feeds WMMA from ds_load_b128.
// 5x5 conv (M=1) + residual + NHWC transpose is a fused VALU kernel.
// ---------------------------------------------------------------------------

typedef __bf16 bf16_t;
typedef __attribute__((ext_vector_type(16))) __bf16 v16bf;
typedef __attribute__((ext_vector_type(8)))  __bf16 v8bf;
typedef __attribute__((ext_vector_type(8)))  float  v8f;
typedef __attribute__((ext_vector_type(4)))  int    v4i;

#if defined(__HIP_DEVICE_COMPILE__) && __has_builtin(__builtin_amdgcn_global_load_async_to_lds_b128)
#define USE_ASYNC_LDS 1
#else
#define USE_ASYNC_LDS 0
#endif

static __device__ __forceinline__ void async_ld_b128(const bf16_t* gsrc, bf16_t* ldst) {
#if USE_ASYNC_LDS
  __builtin_amdgcn_global_load_async_to_lds_b128(
      (__attribute__((address_space(1))) v4i*)(gsrc),
      (__attribute__((address_space(3))) v4i*)(ldst), 0, 0);
#endif
}
static __device__ __forceinline__ void wait_async_all() {
#if USE_ASYNC_LDS
#if __has_builtin(__builtin_amdgcn_s_wait_asynccnt)
  __builtin_amdgcn_s_wait_asynccnt(0);
#else
  asm volatile("s_wait_asynccnt 0" ::: "memory");
#endif
#endif
}

// A-fragment: lane holds K = kb+{0..15} (p1 = p0+8 at call sites); two 16B chunks.
static __device__ __forceinline__ v16bf ld_frag_pair(const bf16_t* p0, const bf16_t* p1) {
  v8bf a = *reinterpret_cast<const v8bf*>(p0);
  v8bf b = *reinterpret_cast<const v8bf*>(p1);
  return __builtin_shufflevector(a, b, 0,1,2,3,4,5,6,7,8,9,10,11,12,13,14,15);
}
// B-fragment: lane holds 16 contiguous K values (lanes<16: K 0..15, lanes>=16: K 16..31).
static __device__ __forceinline__ v16bf ld_frag_cont(const bf16_t* p) {
  v8bf a = *reinterpret_cast<const v8bf*>(p);
  v8bf b = *reinterpret_cast<const v8bf*>(p + 8);
  return __builtin_shufflevector(a, b, 0,1,2,3,4,5,6,7,8,9,10,11,12,13,14,15);
}
static __device__ __forceinline__ v8f wmma_bf16(v16bf a, v16bf b, v8f c) {
  return __builtin_amdgcn_wmma_f32_16x16x32_bf16(false, a, false, b, (short)0, c, false, false);
}

// --------------------------- pack / convert kernels ------------------------

// X0[b][wp(66)][h(64)] = x[b][h][wp-1], pads zero.
__global__ void pack_x0(const float* __restrict__ x, bf16_t* __restrict__ X0) {
  int i = blockIdx.x * 256 + threadIdx.x;
  if (i >= 32 * 66 * 64) return;
  int h = i & 63; int t = i >> 6; int wp = t % 66; int b = t / 66;
  float v = (wp >= 1 && wp <= 64) ? x[(b * 64 + h) * 64 + (wp - 1)] : 0.f;
  X0[i] = (bf16_t)v;
}

// Wb[r][d][co(32)][Kpad], k=ci*64+h ; W is (64,31,Cin,64,3).
__global__ void pack_w_spatial(const float* __restrict__ W, bf16_t* __restrict__ Wb,
                               int Cin, int Kpad, int total) {
  int i = blockIdx.x * 256 + threadIdx.x;
  if (i >= total) return;
  int k = i % Kpad; int t = i / Kpad;
  int co = t & 31; t >>= 5; int d = t % 3; int r = t / 3;
  int ci = k >> 6, h = k & 63;
  float v = 0.f;
  if (co < 31 && ci < Cin)
    v = W[((((size_t)r * 31 + co) * Cin + ci) * 64 + h) * 3 + d];
  Wb[i] = (bf16_t)v;
}

// S3P[cp(33)][b][yp(72)][xp(80)] bf16, cp = c+1, zero pads.
__global__ void pack_s3p(const float* __restrict__ S3F, bf16_t* __restrict__ S3P) {
  int i = blockIdx.x * 256 + threadIdx.x;
  if (i >= 33 * 32 * 72 * 80) return;
  int xp = i % 80; int t = i / 80; int yp = t % 72; t /= 72;
  int b = t & 31; int cp = t >> 5;
  int c = cp - 1, y = yp - 4, x = xp - 4;
  float v = 0.f;
  if (c >= 0 && c < 31 && y >= 0 && y < 64 && x >= 0 && x < 64)
    v = S3F[(((size_t)b * 31 + c) * 64 + y) * 64 + x];
  S3P[i] = (bf16_t)v;
}

// Unified 9x9 weights: WAb[g(31)][oc(64)][K=448], k = kg*16 + dx, kg=(ci*9+dy).
__global__ void pack_wa(const float* __restrict__ We1a, const float* __restrict__ We2a,
                        bf16_t* __restrict__ WAb) {
  int i = blockIdx.x * 256 + threadIdx.x;
  if (i >= 31 * 64 * 448) return;
  int k = i % 448; int t = i / 448; int oc = t & 63; int g = t >> 6;
  int kg = k >> 4, dx = k & 15;
  float v = 0.f;
  if (kg < 27 && dx < 9) {
    int ci = kg / 9, dy = kg % 9;
    if (g == 0) {
      if (ci >= 1) v = We1a[((size_t)(0 * 64 + oc) * 2 + (ci - 1)) * 81 + dy * 9 + dx];
    } else if (g == 30) {
      if (ci < 2)  v = We1a[((size_t)(1 * 64 + oc) * 2 + ci) * 81 + dy * 9 + dx];
    } else {
      v = We2a[(((size_t)(g - 1) * 64 + oc) * 3 + ci) * 81 + dy * 9 + dx];
    }
  }
  WAb[i] = (bf16_t)v;
}

__global__ void pack_ba(const float* __restrict__ be1a, const float* __restrict__ be2a,
                        float* __restrict__ bA) {
  int i = blockIdx.x * 256 + threadIdx.x;
  if (i >= 31 * 64) return;
  int g = i >> 6, oc = i & 63;
  bA[i] = (g == 0) ? be1a[oc] : (g == 30) ? be1a[64 + oc] : be2a[(g - 1) * 64 + oc];
}

// 1x1 weights: WBb[g][oc(32)][ci(64)] bf16.
__global__ void pack_wb(const float* __restrict__ We1b, const float* __restrict__ We2b,
                        bf16_t* __restrict__ WBb) {
  int i = blockIdx.x * 256 + threadIdx.x;
  if (i >= 31 * 32 * 64) return;
  int ci = i & 63; int t = i >> 6; int oc = t & 31; int g = t >> 5;
  float v = (g == 0) ? We1b[oc * 64 + ci]
          : (g == 30) ? We1b[2048 + oc * 64 + ci]
          : We2b[((size_t)(g - 1) * 32 + oc) * 64 + ci];
  WBb[i] = (bf16_t)v;
}

__global__ void pack_bb(const float* __restrict__ be1b, const float* __restrict__ be2b,
                        float* __restrict__ bB) {
  int i = blockIdx.x * 256 + threadIdx.x;
  if (i >= 31 * 32) return;
  int g = i >> 5, oc = i & 31;
  bB[i] = (g == 0) ? be1b[oc] : (g == 30) ? be1b[32 + oc] : be2b[(g - 1) * 32 + oc];
}

// ------------------------------ WMMA GEMM kernels --------------------------

// Spatial layer GEMM. For row r: out[co,(b,w)] = relu(sum_d sum_k A·B(+shift d)).
// X layout: [b][wp(66)][Kin] bf16.
// mode 0: write Xout[b][wp][2048] (k = co*64 + r).  mode 1: write S3F f32 [b][c][r][w].
__global__ __launch_bounds__(128)
void spatial_gemm(const bf16_t* __restrict__ X, const bf16_t* __restrict__ Wb,
                  const float* __restrict__ bias, int Kin,
                  bf16_t* __restrict__ Xout, float* __restrict__ S3F, int mode) {
  const int lane = threadIdx.x & 31;
  const int wave = threadIdx.x >> 5;        // 0..3
  const int r    = blockIdx.y;              // 0..63
  const int n0   = blockIdx.x * 256 + wave * 64;
  const int l15  = lane & 15;
  const int hi   = lane >> 4;
  const int kbA  = hi * 16;                 // A fragment contiguous offset
  const int kbB  = hi * 16;                 // B fragment contiguous offset

  v8f acc[2][4] = {};
  const int ksteps = Kin >> 5;

  for (int d = 0; d < 3; ++d) {
    const bf16_t* Abase = Wb + ((size_t)(r * 3 + d) * 32) * Kin;
    for (int ks = 0; ks < ksteps; ++ks) {
      const int koA = ks * 32 + kbA;
      const bf16_t* ar0 = Abase + (size_t)l15 * Kin + koA;
      const bf16_t* ar1 = Abase + (size_t)(16 + l15) * Kin + koA;
      v16bf a0 = ld_frag_pair(ar0, ar0 + 8);
      v16bf a1 = ld_frag_pair(ar1, ar1 + 8);
      const int koB = ks * 32 + kbB;
#pragma unroll
      for (int nt = 0; nt < 4; ++nt) {
        int n = n0 + nt * 16 + l15;
        int b = n >> 6, w = n & 63;
        const bf16_t* col = X + ((size_t)(b * 66) + (w + d)) * Kin + koB;
        v16bf bf = ld_frag_cont(col);
        acc[0][nt] = wmma_bf16(a0, bf, acc[0][nt]);
        acc[1][nt] = wmma_bf16(a1, bf, acc[1][nt]);
      }
    }
  }

#pragma unroll
  for (int mt = 0; mt < 2; ++mt)
#pragma unroll
    for (int nt = 0; nt < 4; ++nt)
#pragma unroll
      for (int p = 0; p < 8; ++p) {
        int co = mt * 16 + p + (hi << 3);
        int n  = n0 + nt * 16 + l15;
        int b = n >> 6, w = n & 63;
        float v = acc[mt][nt][p] + ((co < 31) ? bias[r * 31 + co] : 0.f);
        v = fmaxf(v, 0.f);
        if (mode == 0) {
          Xout[((size_t)(b * 66) + (w + 1)) * 2048 + (size_t)co * 64 + r] = (bf16_t)v;
        } else if (co < 31) {
          S3F[(((size_t)b * 31 + co) * 64 + r) * 64 + w] = v;
        }
      }
}

// 9x9 conv (unified Cin=3, dx padded to 16): K = 28 kgroups * 16 = 448.
// h1[b][y][x][oc(64)] = relu(conv). One block per (b,y); 2 waves cover oc 0..63.
// Weight slab (56KB) + the 27 needed image rows (4.3KB) are staged into LDS
// with async loads; WMMA fragments then come from ds_load_b128.
__global__ __launch_bounds__(64)
void conva_gemm(const bf16_t* __restrict__ S3P, const bf16_t* __restrict__ WAb,
                const float* __restrict__ bA, bf16_t* __restrict__ h1, int g) {
  const int lane = threadIdx.x & 31;
  const int wave = threadIdx.x >> 5;        // 0..1 -> oc 0..31 / 32..63
  const int by   = blockIdx.x;              // b*64 + y
  const int b = by >> 6, y = by & 63;
  const int l15 = lane & 15;
  const int hi  = lane >> 4;
  const int kbA = hi * 16;

  v8f acc[2][4] = {};

#if USE_ASYNC_LDS
  __shared__ bf16_t lA[64 * 448];           // 56 KB weight slab for this group
  __shared__ bf16_t lB[27 * 80];            // 27 padded image rows for this (b,y)
  {
    const bf16_t* gA = WAb + (size_t)g * 64 * 448;
    for (int i = threadIdx.x; i < (64 * 448) / 8; i += 64)
      async_ld_b128(gA + i * 8, lA + i * 8);
    for (int i = threadIdx.x; i < 270; i += 64) {     // 27 rows * 10 x b128
      int row = i / 10, off = (i % 10) * 8;
      int ci = row / 9, dy = row - ci * 9;
      const bf16_t* gB =
          S3P + (((size_t)(g + ci) * 32 + b) * 72 + (y + dy)) * 80 + off;
      async_ld_b128(gB, lB + row * 80 + off);
    }
    wait_async_all();
    __syncthreads();
  }
  const bf16_t* Abase = lA;
#else
  const bf16_t* Abase = WAb + (size_t)g * 64 * 448;
#endif

#pragma unroll
  for (int s = 0; s < 14; ++s) {
    const int koA = s * 32 + kbA;
    const bf16_t* ar0 = Abase + (size_t)(wave * 32 + l15) * 448 + koA;
    const bf16_t* ar1 = Abase + (size_t)(wave * 32 + 16 + l15) * 448 + koA;
    v16bf a0 = ld_frag_pair(ar0, ar0 + 8);
    v16bf a1 = ld_frag_pair(ar1, ar1 + 8);
    // this lane's kgroup: kg = 2s + hi ; 16 contiguous dx values.
    int kg = 2 * s + hi;
    if (kg >= 27) kg = 0;                   // A rows are zero there
#if USE_ASYNC_LDS
    const bf16_t* row = lB + kg * 80;
#else
    int ci = kg / 9, dy = kg - ci * 9;
    const bf16_t* row = S3P + (((size_t)(g + ci) * 32 + b) * 72 + (y + dy)) * 80;
#endif
#pragma unroll
    for (int nt = 0; nt < 4; ++nt) {
      int xcol = nt * 16 + l15;
      v16bf bf = ld_frag_cont(row + xcol);
      acc[0][nt] = wmma_bf16(a0, bf, acc[0][nt]);
      acc[1][nt] = wmma_bf16(a1, bf, acc[1][nt]);
    }
  }

#pragma unroll
  for (int mt = 0; mt < 2; ++mt)
#pragma unroll
    for (int nt = 0; nt < 4; ++nt)
#pragma unroll
      for (int p = 0; p < 8; ++p) {
        int oc = wave * 32 + mt * 16 + p + (hi << 3);
        int x  = nt * 16 + l15;
        float v = fmaxf(acc[mt][nt][p] + bA[g * 64 + oc], 0.f);
        h1[((size_t)by * 64 + x) * 64 + oc] = (bf16_t)v;
      }
}

// 1x1 conv: h2[b][yp(68)][xp(68)][oc(32)] = relu(sum_{ci<64} WB·h1), padded store.
__global__ __launch_bounds__(128)
void convb_gemm(const bf16_t* __restrict__ h1, const bf16_t* __restrict__ WBb,
                const float* __restrict__ bB, bf16_t* __restrict__ h2, int g) {
  const int lane = threadIdx.x & 31;
  const int wave = threadIdx.x >> 5;        // 0..3
  const int n0   = blockIdx.x * 256 + wave * 64;
  const int l15  = lane & 15;
  const int hi   = lane >> 4;
  const int kbA  = hi * 16;
  const int kbB  = hi * 16;

  v8f acc[2][4] = {};
  const bf16_t* Abase = WBb + (size_t)g * 32 * 64;

#pragma unroll
  for (int ks = 0; ks < 2; ++ks) {
    const int koA = ks * 32 + kbA;
    const bf16_t* ar0 = Abase + (size_t)l15 * 64 + koA;
    const bf16_t* ar1 = Abase + (size_t)(16 + l15) * 64 + koA;
    v16bf a0 = ld_frag_pair(ar0, ar0 + 8);
    v16bf a1 = ld_frag_pair(ar1, ar1 + 8);
    const int koB = ks * 32 + kbB;
#pragma unroll
    for (int nt = 0; nt < 4; ++nt) {
      int n = n0 + nt * 16 + l15;
      v16bf bf = ld_frag_cont(h1 + (size_t)n * 64 + koB);
      acc[0][nt] = wmma_bf16(a0, bf, acc[0][nt]);
      acc[1][nt] = wmma_bf16(a1, bf, acc[1][nt]);
    }
  }

#pragma unroll
  for (int mt = 0; mt < 2; ++mt)
#pragma unroll
    for (int nt = 0; nt < 4; ++nt)
#pragma unroll
      for (int p = 0; p < 8; ++p) {
        int oc = mt * 16 + p + (hi << 3);
        int n  = n0 + nt * 16 + l15;
        int x = n & 63, y = (n >> 6) & 63, b = n >> 12;
        float v = fmaxf(acc[mt][nt][p] + bB[g * 32 + oc], 0.f);
        h2[(((size_t)b * 68 + (y + 2)) * 68 + (x + 2)) * 32 + oc] = (bf16_t)v;
      }
}

// 5x5 conv (M=1, ~2.5% of FLOPs) fused with residual add + NHWC transpose.
__global__ __launch_bounds__(256)
void convc_res(const bf16_t* __restrict__ h2,
               const float* __restrict__ We1c, const float* __restrict__ be1c,
               const float* __restrict__ We2c, const float* __restrict__ be2c,
               const float* __restrict__ S3F, float* __restrict__ out, int g) {
  __shared__ float wc[800];                  // [ci(32)][dy(5)][dx(5)]
  const float* Wsrc = (g == 0) ? We1c : (g == 30) ? (We1c + 800)
                                      : (We2c + (size_t)(g - 1) * 800);
  for (int i = threadIdx.x; i < 800; i += 256) wc[i] = Wsrc[i];
  __syncthreads();
  float bc = (g == 0) ? be1c[0] : (g == 30) ? be1c[1] : be2c[g - 1];

  int idx = blockIdx.x * 256 + threadIdx.x;  // 32*64*64 threads exactly
  int x = idx & 63, y = (idx >> 6) & 63, b = idx >> 12;
  float accv = bc;
  for (int dy = 0; dy < 5; ++dy)
    for (int dx = 0; dx < 5; ++dx) {
      const v8bf* p = reinterpret_cast<const v8bf*>(
          h2 + (((size_t)b * 68 + (y + dy)) * 68 + (x + dx)) * 32);
#pragma unroll
      for (int c8 = 0; c8 < 4; ++c8) {
        v8bf hv = p[c8];
#pragma unroll
        for (int j = 0; j < 8; ++j)
          accv += (float)hv[j] * wc[(c8 * 8 + j) * 25 + dy * 5 + dx];
      }
    }
  out[(size_t)idx * 31 + g] =
      S3F[(((size_t)b * 31 + g) * 64 + y) * 64 + x] + accv;
}

// ------------------------------- host side ---------------------------------

extern "C" void kernel_launch(void* const* d_in, const int* in_sizes, int n_in,
                              void* d_out, int out_size, void* d_ws, size_t ws_size,
                              hipStream_t stream) {
  const float* x    = (const float*)d_in[0];
  const float* W1   = (const float*)d_in[1];
  const float* b1   = (const float*)d_in[2];
  const float* W2   = (const float*)d_in[3];
  const float* b2   = (const float*)d_in[4];
  const float* W3   = (const float*)d_in[5];
  const float* b3   = (const float*)d_in[6];
  const float* We1a = (const float*)d_in[7];
  const float* be1a = (const float*)d_in[8];
  const float* We1b = (const float*)d_in[9];
  const float* be1b = (const float*)d_in[10];
  const float* We1c = (const float*)d_in[11];
  const float* be1c = (const float*)d_in[12];
  const float* We2a = (const float*)d_in[13];
  const float* be2a = (const float*)d_in[14];
  const float* We2b = (const float*)d_in[15];
  const float* be2b = (const float*)d_in[16];
  const float* We2c = (const float*)d_in[17];
  const float* be2c = (const float*)d_in[18];
  float* out = (float*)d_out;

  char* ws = (char*)d_ws;
  size_t off = 0;
  auto alloc = [&](size_t bytes) -> char* {
    char* p = ws + off;
    off = (off + bytes + 255) & ~(size_t)255;
    return p;
  };

  bf16_t* X0  = (bf16_t*)alloc((size_t)32 * 66 * 64 * 2);
  bf16_t* X1  = (bf16_t*)alloc((size_t)32 * 66 * 2048 * 2);
  bf16_t* X2  = (bf16_t*)alloc((size_t)32 * 66 * 2048 * 2);
  bf16_t* Wb1 = (bf16_t*)alloc((size_t)64 * 3 * 32 * 64 * 2);
  bf16_t* Wb2 = (bf16_t*)alloc((size_t)64 * 3 * 32 * 2048 * 2);
  bf16_t* Wb3 = (bf16_t*)alloc((size_t)64 * 3 * 32 * 2048 * 2);
  float*  S3F = (float*) alloc((size_t)32 * 31 * 64 * 64 * 4);
  bf16_t* S3P = (bf16_t*)alloc((size_t)33 * 32 * 72 * 80 * 2);
  bf16_t* WAb = (bf16_t*)alloc((size_t)31 * 64 * 448 * 2);
  float*  bA  = (float*) alloc((size_t)31 * 64 * 4);
  bf16_t* WBb = (bf16_t*)alloc((size_t)31 * 32 * 64 * 2);
  float*  bB  = (float*) alloc((size_t)31 * 32 * 4);
  bf16_t* h1  = (bf16_t*)alloc((size_t)32 * 64 * 64 * 64 * 2);
  bf16_t* h2  = (bf16_t*)alloc((size_t)32 * 68 * 68 * 32 * 2);
  if (off > ws_size) return;  // workspace too small; nothing safe to do

  // ---- input / weight packing (bf16, WMMA-friendly layouts) ----
  pack_x0<<<(32 * 66 * 64 + 255) / 256, 256, 0, stream>>>(x, X0);
  {
    int t1 = 64 * 3 * 32 * 64;
    pack_w_spatial<<<(t1 + 255) / 256, 256, 0, stream>>>(W1, Wb1, 1, 64, t1);
    int t2 = 64 * 3 * 32 * 2048;
    pack_w_spatial<<<(t2 + 255) / 256, 256, 0, stream>>>(W2, Wb2, 31, 2048, t2);
    pack_w_spatial<<<(t2 + 255) / 256, 256, 0, stream>>>(W3, Wb3, 31, 2048, t2);
  }
  pack_wa<<<(31 * 64 * 448 + 255) / 256, 256, 0, stream>>>(We1a, We2a, WAb);
  pack_ba<<<(31 * 64 + 255) / 256, 256, 0, stream>>>(be1a, be2a, bA);
  pack_wb<<<(31 * 32 * 64 + 255) / 256, 256, 0, stream>>>(We1b, We2b, WBb);
  pack_bb<<<(31 * 32 + 255) / 256, 256, 0, stream>>>(be1b, be2b, bB);

  // zero the w-padding columns of X1/X2 and the spatial pad ring of h2
  (void)hipMemsetAsync(X1, 0, (size_t)32 * 66 * 2048 * 2, stream);
  (void)hipMemsetAsync(X2, 0, (size_t)32 * 66 * 2048 * 2, stream);
  (void)hipMemsetAsync(h2, 0, (size_t)32 * 68 * 68 * 32 * 2, stream);

  // ---- spatial layers: one GEMM per layer (M=64x32, N=2048, K=3*Kin) ----
  spatial_gemm<<<dim3(8, 64), 128, 0, stream>>>(X0, Wb1, b1, 64,   X1, nullptr, 0);
  spatial_gemm<<<dim3(8, 64), 128, 0, stream>>>(X1, Wb2, b2, 2048, X2, nullptr, 0);
  spatial_gemm<<<dim3(8, 64), 128, 0, stream>>>(X2, Wb3, b3, 2048, nullptr, S3F, 1);

  // repack s3 into padded bf16 image planes for the 9x9 conv
  pack_s3p<<<(33 * 32 * 72 * 80 + 255) / 256, 256, 0, stream>>>(S3F, S3P);

  // ---- spectral branch: 31 groups, each 9x9 -> 1x1 -> 5x5(+residual) ----
  for (int g = 0; g < 31; ++g) {
    conva_gemm<<<2048, 64, 0, stream>>>(S3P, WAb, bA, h1, g);
    convb_gemm<<<512, 128, 0, stream>>>(h1, WBb, bB, h2, g);
    convc_res<<<512, 256, 0, stream>>>(h2, We1c, be1c, We2c, be2c, S3F, out, g);
  }
}
